// Pointer_46059229282787
// MI455X (gfx1250) — matrix operand order repaired and checked by reading, not compiled, not run on hardware
//
#include <hip/hip_runtime.h>
#include <hip/hip_bf16.h>

typedef __attribute__((ext_vector_type(16))) _Float16 v16h;
typedef __attribute__((ext_vector_type(8)))  _Float16 v8h;
typedef __attribute__((ext_vector_type(8)))  float    v8f;

#define B_  128
#define N_  512
#define H_  512
#define AS_STRIDE 520   // padded f16 row stride (breaks LDS bank conflicts)
#define BS_STRIDE 520
#define A_BYTES   (128 * AS_STRIDE * 2)      // 133120
#define BSLAB     (16 * BS_STRIDE * 2)       // 16640 bytes per 16-row W slab

__device__ __forceinline__ v16h cat8(v8h lo, v8h hi) {
  return __builtin_shufflevector(lo, hi, 0,1,2,3,4,5,6,7,8,9,10,11,12,13,14,15);
}
__device__ __forceinline__ float sigmoidf_(float x) { return 1.0f / (1.0f + __expf(-x)); }

// CDNA5 async global->LDS copy (ASYNCcnt tracked); dst = 32-bit LDS byte offset.
__device__ __forceinline__ void async_ld16(unsigned lds_off, const void* g) {
  asm volatile("global_load_async_to_lds_b128 %0, %1, off"
               :: "v"(lds_off), "v"((unsigned long long)(uintptr_t)g) : "memory");
}
__device__ __forceinline__ void wait_async0() {
  asm volatile("s_wait_asynccnt 0x0" ::: "memory");
}

// ---------------------------------------------------------------------------
// K0: convert static parts of enc_W, tgt_W, ptr_W to f16 (row-major h x k)
// ---------------------------------------------------------------------------
__global__ void convert_w_kernel(const float* __restrict__ encW,
                                 const float* __restrict__ tgtW,
                                 const float* __restrict__ ptrW,
                                 _Float16* __restrict__ w16) {
  int idx = blockIdx.x * 256 + threadIdx.x;        // 3 * 512*512 = 786432 total
  int which = idx >> 18;
  int r = idx & 262143;
  int h = r >> 9, k = r & 511;
  float v;
  if      (which == 0) v = encW[(size_t)h * 1024 + k];
  else if (which == 1) v = tgtW[(size_t)h * 1024 + k];
  else                 v = ptrW[(size_t)h * 1536 + k];
  w16[idx] = (_Float16)v;
}

// ---------------------------------------------------------------------------
// K1: GRU step.  One block per batch row.
// ---------------------------------------------------------------------------
__global__ void gru_kernel(const float* __restrict__ dec_hidden,
                           const float* __restrict__ last_hh,
                           const float* __restrict__ w_ih,
                           const float* __restrict__ w_hh,
                           const float* __restrict__ b_ih,
                           const float* __restrict__ b_hh,
                           float* __restrict__ h_new,
                           float* __restrict__ out_hh) {
  int b = blockIdx.x, t = threadIdx.x;
  __shared__ float xs[512], hs[512], gx[1536], gh[1536];
  xs[t]       = dec_hidden[(b << 9) + t];
  xs[t + 256] = dec_hidden[(b << 9) + t + 256];
  hs[t]       = last_hh[(b << 9) + t];
  hs[t + 256] = last_hh[(b << 9) + t + 256];
  __syncthreads();
  for (int j = t; j < 1536; j += 256) {
    float sx = b_ih[j], sh = b_hh[j];
    const float* wi = w_ih + (size_t)j * 512;
    const float* wh = w_hh + (size_t)j * 512;
    #pragma unroll 4
    for (int k = 0; k < 512; ++k) { sx += wi[k] * xs[k]; sh += wh[k] * hs[k]; }
    gx[j] = sx; gh[j] = sh;
  }
  __syncthreads();
  #pragma unroll
  for (int r = 0; r < 2; ++r) {
    int i = t + (r << 8);
    float rg = sigmoidf_(gx[i] + gh[i]);
    float z  = sigmoidf_(gx[512 + i] + gh[512 + i]);
    float n  = tanhf(gx[1024 + i] + rg * gh[1024 + i]);
    float hv = (1.0f - z) * n + z * hs[i];
    h_new[(b << 9) + i] = hv;
    out_hh[(b << 9) + i] = hv;
  }
}

// ---------------------------------------------------------------------------
// K2: per-batch dynamic vectors d_enc, d_tgt, g_state
// ---------------------------------------------------------------------------
__global__ void dynvec_kernel(const float* __restrict__ encW,
                              const float* __restrict__ tgtW,
                              const float* __restrict__ ptrW,
                              const float* __restrict__ hnew,
                              const float* __restrict__ target_hidden,
                              const float* __restrict__ state_emb,
                              float* __restrict__ d_enc,
                              float* __restrict__ d_tgt,
                              float* __restrict__ g_state) {
  int b = blockIdx.x, t = threadIdx.x;
  __shared__ float hn[512], th[512], se[512];
  hn[t] = hnew[(b << 9) + t];            hn[t + 256] = hnew[(b << 9) + t + 256];
  th[t] = target_hidden[(b << 9) + t];   th[t + 256] = target_hidden[(b << 9) + t + 256];
  se[t] = state_emb[(b << 9) + t];       se[t + 256] = state_emb[(b << 9) + t + 256];
  __syncthreads();
  #pragma unroll
  for (int r = 0; r < 2; ++r) {
    int h = t + (r << 8);
    const float* we = encW + (size_t)h * 1024 + 512;
    const float* wt = tgtW + (size_t)h * 1024 + 512;
    const float* wp = ptrW + (size_t)h * 1536 + 1024;
    float s1 = 0.f, s2 = 0.f, s3 = 0.f;
    #pragma unroll 4
    for (int k = 0; k < 512; ++k) {
      s1 += we[k] * hn[k]; s2 += wt[k] * th[k]; s3 += wp[k] * se[k];
    }
    d_enc[(b << 9) + h] = s1; d_tgt[(b << 9) + h] = s2; g_state[(b << 9) + h] = s3;
  }
}

// ---------------------------------------------------------------------------
// K3: WMMA pass 1 — enc & tgt attention scores.
//  grid = 512 workgroups (128 rows each), 256 threads (8 waves).
//  B slabs staged with global_load_async_to_lds_b128, double buffered.
// ---------------------------------------------------------------------------
__global__ void __launch_bounds__(256)
pass1_kernel(const float* __restrict__ static_hidden,
             const _Float16* __restrict__ We16,
             const _Float16* __restrict__ Wt16,
             const float* __restrict__ d_enc,
             const float* __restrict__ d_tgt,
             const float* __restrict__ enc_v,
             const float* __restrict__ tgt_v,
             float* __restrict__ scores_enc,
             float* __restrict__ scores_tgt) {
  extern __shared__ char smem[];
  _Float16* As = (_Float16*)smem;                       // 128 x AS_STRIDE
  _Float16* B0 = (_Float16*)(smem + A_BYTES);           // buf0: Be | Bt
  _Float16* B1 = (_Float16*)(smem + A_BYTES + 2 * BSLAB);

  int t = threadIdx.x;
  int r0 = blockIdx.x * 128;
  int b  = r0 >> 9;

  // LDS byte offsets for async destinations (generic ptr low 32 bits = LDS offset)
  unsigned beOff[2] = { (unsigned)(uintptr_t)B0, (unsigned)(uintptr_t)B1 };

  // helper lambda: issue async copies of both 16KB W slabs for h-tile ht into buf
  auto stage = [&](int ht, int buf) {
    const char* gse = (const char*)(We16 + (size_t)ht * 16 * 512);
    const char* gst = (const char*)(Wt16 + (size_t)ht * 16 * 512);
    #pragma unroll
    for (int i = 0; i < 4; ++i) {
      int cid = t + i * 256;                 // 1024 chunks of 16B per slab
      int j = cid >> 6, kc = cid & 63;       // 64 chunks per W row
      unsigned off = (unsigned)(j * (BS_STRIDE * 2) + kc * 16);
      async_ld16(beOff[buf] + off,          gse + (size_t)cid * 16);
      async_ld16(beOff[buf] + BSLAB + off,  gst + (size_t)cid * 16);
    }
  };

  // prologue: start slab 0 while we stage the A tile through VGPRs (needs cvt)
  stage(0, 0);

  const float4* src = (const float4*)(static_hidden + (size_t)r0 * 512);
  #pragma unroll 4
  for (int i = 0; i < 64; ++i) {
    int idx4 = t + i * 256;
    int flat = idx4 << 2;
    int row = flat >> 9, k = flat & 511;
    float4 v = src[idx4];
    _Float16* dst = As + row * AS_STRIDE + k;
    dst[0] = (_Float16)v.x; dst[1] = (_Float16)v.y;
    dst[2] = (_Float16)v.z; dst[3] = (_Float16)v.w;
  }

  int wave = t >> 5, lane = t & 31, c = lane & 15, halfk = lane >> 4;
  const _Float16* arow = As + (wave * 16 + c) * AS_STRIDE;
  float sum_e[8] = {0,0,0,0,0,0,0,0};
  float sum_t[8] = {0,0,0,0,0,0,0,0};

  for (int htile = 0; htile < 32; ++htile) {
    int cur = htile & 1;
    wait_async0();            // this wave's slab copies into buf[cur] are done
    __syncthreads();          // everyone's copies done; prev reads of buf[cur^1] done
    if (htile + 1 < 32) stage(htile + 1, cur ^ 1);

    const _Float16* Bb = cur ? B1 : B0;
    const _Float16* be_row = Bb + c * BS_STRIDE + 16 * halfk;
    const _Float16* bt_row = be_row + 16 * BS_STRIDE;

    v8f acc_e = {}; v8f acc_t = {};
    #pragma unroll
    for (int kb = 0; kb < 16; ++kb) {
      int k0 = kb * 32;
      v16h a  = cat8(*(const v8h*)(arow + k0 + 16 * halfk),
                     *(const v8h*)(arow + k0 + 16 * halfk + 8));
      v16h be = cat8(*(const v8h*)(be_row + k0), *(const v8h*)(be_row + k0 + 8));
      v16h bt = cat8(*(const v8h*)(bt_row + k0), *(const v8h*)(bt_row + k0 + 8));
      acc_e = __builtin_amdgcn_wmma_f32_16x16x32_f16(false, a, false, be, (short)0, acc_e, false, false);
      acc_t = __builtin_amdgcn_wmma_f32_16x16x32_f16(false, a, false, bt, (short)0, acc_t, false, false);
    }
    int hc = htile * 16 + c;
    float de = d_enc[(b << 9) + hc], ve = enc_v[hc];
    float dt = d_tgt[(b << 9) + hc], vt = tgt_v[hc];
    #pragma unroll
    for (int i = 0; i < 8; ++i) {
      sum_e[i] += tanhf(acc_e[i] + de) * ve;
      sum_t[i] += tanhf(acc_t[i] + dt) * vt;
    }
  }

  // butterfly reduce over the 16 lanes that share a row group
  #pragma unroll
  for (int m = 8; m >= 1; m >>= 1) {
    #pragma unroll
    for (int i = 0; i < 8; ++i) {
      sum_e[i] += __shfl_xor(sum_e[i], m, 32);
      sum_t[i] += __shfl_xor(sum_t[i], m, 32);
    }
  }
  if (c == 0) {
    int rr = r0 + wave * 16 + halfk * 8;
    #pragma unroll
    for (int i = 0; i < 8; ++i) {
      scores_enc[rr + i] = sum_e[i];
      scores_tgt[rr + i] = sum_t[i];
    }
  }
}

// ---------------------------------------------------------------------------
// K4: softmax over N, then context = attn @ static_hidden.
//  grid = 256 blocks (first 128 enc, next 128 tgt).
// ---------------------------------------------------------------------------
__global__ void softmax_ctx_kernel(const float* __restrict__ scores_enc,
                                   const float* __restrict__ scores_tgt,
                                   const float* __restrict__ static_hidden,
                                   float* __restrict__ context,
                                   float* __restrict__ context_t) {
  int blk = blockIdx.x, which = blk >> 7, b = blk & 127, t = threadIdx.x;
  const float* s = which ? scores_tgt : scores_enc;
  float* o       = which ? context_t  : context;
  __shared__ float w[512];
  __shared__ float red[256];
  w[t]       = s[(b << 9) + t];
  w[t + 256] = s[(b << 9) + t + 256];
  __syncthreads();
  red[t] = fmaxf(w[t], w[t + 256]);
  __syncthreads();
  for (int st = 128; st > 0; st >>= 1) {
    if (t < st) red[t] = fmaxf(red[t], red[t + st]);
    __syncthreads();
  }
  float m = red[0];
  __syncthreads();
  float e0 = __expf(w[t] - m), e1 = __expf(w[t + 256] - m);
  red[t] = e0 + e1;
  __syncthreads();
  for (int st = 128; st > 0; st >>= 1) {
    if (t < st) red[t] += red[t + st];
    __syncthreads();
  }
  float inv = 1.0f / red[0];
  __syncthreads();
  w[t] = e0 * inv; w[t + 256] = e1 * inv;
  __syncthreads();
  #pragma unroll
  for (int r = 0; r < 2; ++r) {
    int h = t + (r << 8);
    float acc = 0.f;
    const float* sh = static_hidden + ((size_t)b << 18) + h;
    #pragma unroll 4
    for (int n = 0; n < 512; ++n) acc += w[n] * sh[(size_t)n << 9];
    o[(b << 9) + h] = acc;
  }
}

// ---------------------------------------------------------------------------
// K5: c1 = ptr_W[:,H:2H] @ context ; c1t = ptr_W[:,H:2H] @ context_t
// ---------------------------------------------------------------------------
__global__ void cvec_kernel(const float* __restrict__ ptrW,
                            const float* __restrict__ ctx,
                            const float* __restrict__ ctx_t,
                            float* __restrict__ c1,
                            float* __restrict__ c1t) {
  int b = blockIdx.x, t = threadIdx.x;
  __shared__ float ca[512], cb[512];
  ca[t] = ctx[(b << 9) + t];    ca[t + 256] = ctx[(b << 9) + t + 256];
  cb[t] = ctx_t[(b << 9) + t];  cb[t + 256] = ctx_t[(b << 9) + t + 256];
  __syncthreads();
  #pragma unroll
  for (int r = 0; r < 2; ++r) {
    int h = t + (r << 8);
    const float* wp = ptrW + (size_t)h * 1536 + 512;
    float s1 = 0.f, s2 = 0.f;
    #pragma unroll 4
    for (int k = 0; k < 512; ++k) { s1 += wp[k] * ca[k]; s2 += wp[k] * cb[k]; }
    c1[(b << 9) + h] = s1; c1t[(b << 9) + h] = s2;
  }
}

// ---------------------------------------------------------------------------
// K6: WMMA pass 2 — pointer scores with both contexts; writes final probs.
// ---------------------------------------------------------------------------
__global__ void __launch_bounds__(256)
pass2_kernel(const float* __restrict__ static_hidden,
             const _Float16* __restrict__ Wp16,
             const float* __restrict__ c1,
             const float* __restrict__ c1t,
             const float* __restrict__ g_state,
             const float* __restrict__ ptr_v,
             float* __restrict__ probs_out) {
  extern __shared__ char smem[];
  _Float16* As = (_Float16*)smem;
  _Float16* B0 = (_Float16*)(smem + A_BYTES);
  _Float16* B1 = (_Float16*)(smem + A_BYTES + BSLAB);

  int t = threadIdx.x;
  int r0 = blockIdx.x * 128;
  int b  = r0 >> 9;

  unsigned bOff[2] = { (unsigned)(uintptr_t)B0, (unsigned)(uintptr_t)B1 };

  auto stage = [&](int ht, int buf) {
    const char* gsp = (const char*)(Wp16 + (size_t)ht * 16 * 512);
    #pragma unroll
    for (int i = 0; i < 4; ++i) {
      int cid = t + i * 256;
      int j = cid >> 6, kc = cid & 63;
      unsigned off = (unsigned)(j * (BS_STRIDE * 2) + kc * 16);
      async_ld16(bOff[buf] + off, gsp + (size_t)cid * 16);
    }
  };

  stage(0, 0);

  const float4* src = (const float4*)(static_hidden + (size_t)r0 * 512);
  #pragma unroll 4
  for (int i = 0; i < 64; ++i) {
    int idx4 = t + i * 256;
    int flat = idx4 << 2;
    int row = flat >> 9, k = flat & 511;
    float4 v = src[idx4];
    _Float16* dst = As + row * AS_STRIDE + k;
    dst[0] = (_Float16)v.x; dst[1] = (_Float16)v.y;
    dst[2] = (_Float16)v.z; dst[3] = (_Float16)v.w;
  }

  int wave = t >> 5, lane = t & 31, c = lane & 15, halfk = lane >> 4;
  const _Float16* arow = As + (wave * 16 + c) * AS_STRIDE;
  float sum1[8] = {0,0,0,0,0,0,0,0};
  float sum2[8] = {0,0,0,0,0,0,0,0};

  for (int htile = 0; htile < 32; ++htile) {
    int cur = htile & 1;
    wait_async0();
    __syncthreads();
    if (htile + 1 < 32) stage(htile + 1, cur ^ 1);

    const _Float16* Bb = cur ? B1 : B0;
    const _Float16* bp_row = Bb + c * BS_STRIDE + 16 * halfk;

    v8f acc = {};
    #pragma unroll
    for (int kb = 0; kb < 16; ++kb) {
      int k0 = kb * 32;
      v16h a  = cat8(*(const v8h*)(arow + k0 + 16 * halfk),
                     *(const v8h*)(arow + k0 + 16 * halfk + 8));
      v16h bp = cat8(*(const v8h*)(bp_row + k0), *(const v8h*)(bp_row + k0 + 8));
      acc = __builtin_amdgcn_wmma_f32_16x16x32_f16(false, a, false, bp, (short)0, acc, false, false);
    }
    int hc = htile * 16 + c;
    float gs = g_state[(b << 9) + hc];
    float a1 = c1[(b << 9) + hc] + gs;
    float a2 = c1t[(b << 9) + hc] + gs;
    float vp = ptr_v[hc];
    #pragma unroll
    for (int i = 0; i < 8; ++i) {
      sum1[i] += tanhf(acc[i] + a1) * vp;
      sum2[i] += tanhf(acc[i] + a2) * vp;
    }
  }

  #pragma unroll
  for (int m = 8; m >= 1; m >>= 1) {
    #pragma unroll
    for (int i = 0; i < 8; ++i) {
      sum1[i] += __shfl_xor(sum1[i], m, 32);
      sum2[i] += __shfl_xor(sum2[i], m, 32);
    }
  }
  if (c == 0) {
    int rr = r0 + wave * 16 + halfk * 8;
    #pragma unroll
    for (int i = 0; i < 8; ++i)
      probs_out[rr + i] = 5.0f * sum1[i] + sum2[i];
  }
}

// ---------------------------------------------------------------------------
extern "C" void kernel_launch(void* const* d_in, const int* in_sizes, int n_in,
                              void* d_out, int out_size, void* d_ws, size_t ws_size,
                              hipStream_t stream) {
  const float* static_hidden = (const float*)d_in[0];
  const float* state_emb     = (const float*)d_in[1];
  const float* dec_hidden    = (const float*)d_in[2];
  const float* target_hidden = (const float*)d_in[3];
  const float* last_hh       = (const float*)d_in[4];
  const float* ptr_v         = (const float*)d_in[5];
  const float* ptr_W         = (const float*)d_in[6];
  const float* enc_v         = (const float*)d_in[7];
  const float* enc_W         = (const float*)d_in[8];
  const float* tgt_v         = (const float*)d_in[9];
  const float* tgt_W         = (const float*)d_in[10];
  const float* gw_ih         = (const float*)d_in[11];
  const float* gw_hh         = (const float*)d_in[12];
  const float* gb_ih         = (const float*)d_in[13];
  const float* gb_hh         = (const float*)d_in[14];

  float* out = (float*)d_out;            // [0,65536) probs, [65536,131072) new_last_hh
  float* ws  = (float*)d_ws;

  float* h_new      = ws;                 // 65536 each
  float* d_enc      = ws + 65536;
  float* d_tgt      = ws + 131072;
  float* g_state    = ws + 196608;
  float* c1         = ws + 262144;
  float* c1t        = ws + 327680;
  float* context    = ws + 393216;
  float* context_t  = ws + 458752;
  float* scores_enc = ws + 524288;
  float* scores_tgt = ws + 589824;
  _Float16* w16     = (_Float16*)(ws + 655360);
  _Float16* We16 = w16;
  _Float16* Wt16 = w16 + 262144;
  _Float16* Wp16 = w16 + 524288;

  convert_w_kernel<<<3072, 256, 0, stream>>>(enc_W, tgt_W, ptr_W, w16);
  gru_kernel<<<128, 256, 0, stream>>>(dec_hidden, last_hh, gw_ih, gw_hh, gb_ih, gb_hh,
                                      h_new, out + 65536);
  dynvec_kernel<<<128, 256, 0, stream>>>(enc_W, tgt_W, ptr_W, h_new, target_hidden,
                                         state_emb, d_enc, d_tgt, g_state);

  size_t sh1 = (size_t)A_BYTES + 4 * BSLAB;   // A + 2 double-buffered (enc+tgt) slabs
  pass1_kernel<<<512, 256, sh1, stream>>>(static_hidden, We16, Wt16, d_enc, d_tgt,
                                          enc_v, tgt_v, scores_enc, scores_tgt);

  softmax_ctx_kernel<<<256, 256, 0, stream>>>(scores_enc, scores_tgt, static_hidden,
                                              context, context_t);
  cvec_kernel<<<128, 256, 0, stream>>>(ptr_W, context, context_t, c1, c1t);

  size_t sh2 = (size_t)A_BYTES + 2 * BSLAB;   // A + double-buffered ptr slab
  pass2_kernel<<<512, 256, sh2, stream>>>(static_hidden, Wp16, c1, c1t, g_state,
                                          ptr_v, out);
}
